// SingleHeadAttention_19799799235272
// MI455X (gfx1250) — matrix operand order repaired and checked by reading, not compiled, not run on hardware
//
#include <hip/hip_runtime.h>
#include <math.h>

typedef __attribute__((ext_vector_type(16))) _Float16 v16h;
typedef __attribute__((ext_vector_type(8)))  _Float16 v8h;
typedef __attribute__((ext_vector_type(8)))  float    v8f;
typedef __attribute__((ext_vector_type(4)))  float    v4f;

#define S_LEN   4096
#define DMODEL  1024
#define DHEAD   64
#define NROWS   (4 * 4096)   // B*S flattened rows

// ---------------------------------------------------------------- helpers
__device__ __forceinline__ v16h cat16(v8h lo, v8h hi) {
  return __builtin_shufflevector(lo, hi, 0,1,2,3,4,5,6,7,8,9,10,11,12,13,14,15);
}

__device__ __forceinline__ v8f wmma_f16(v16h a, v16h b, v8f c) {
  // D = A(16x32 f16) * B(32x16 f16) + C(16x16 f32)
  return __builtin_amdgcn_wmma_f32_16x16x32_f16(
      /*neg_a=*/false, a, /*neg_b=*/false, b,
      /*c_mod=*/(short)0, c, /*reuse_a=*/false, /*reuse_b=*/false);
}

// reduce across the 16-lane N-group of the WMMA C layout (xor 1,2,4,8 stays
// inside each 16-lane half of the wave32)
__device__ __forceinline__ float row_max16(float v) {
  v = fmaxf(v, __shfl_xor(v, 1, 32));
  v = fmaxf(v, __shfl_xor(v, 2, 32));
  v = fmaxf(v, __shfl_xor(v, 4, 32));
  v = fmaxf(v, __shfl_xor(v, 8, 32));
  return v;
}
__device__ __forceinline__ float row_sum16(float v) {
  v += __shfl_xor(v, 1, 32);
  v += __shfl_xor(v, 2, 32);
  v += __shfl_xor(v, 4, 32);
  v += __shfl_xor(v, 8, 32);
  return v;
}

// ---------------------------------------------------------------- kernel 1
// Q = x*Wq, K = x*Wk (row-major f16 [NROWS,64]); V stored transposed per
// batch: Vt[b][h][s] f16, so the P*V WMMA B-operand is a contiguous run.
// Weight chunks are staged in LDS pre-transposed into the WMMA B layout;
// B operands are loaded in groups of 4 so WMMAs issue back-to-back.
__global__ __launch_bounds__(256)
void proj_qkv_kernel(const float* __restrict__ x,
                     const float* __restrict__ Wq,
                     const float* __restrict__ Wk,
                     const float* __restrict__ Wv,
                     _Float16* __restrict__ qh,
                     _Float16* __restrict__ kh,
                     _Float16* __restrict__ vt)
{
  // whT[w][n][k]: column n of W, k within the 32-deep chunk (12KB)
  __shared__ _Float16 whT[3][64][32];

  const int tid  = threadIdx.x;
  const int wave = tid >> 5;
  const int lane = tid & 31;
  const int hlf  = lane >> 4;          // which 16-lane half of the wave
  const int ln   = lane & 15;
  const int rowBase = blockIdx.x * 128 + wave * 16;   // 16 rows per wave

  const float* W[3] = {Wq, Wk, Wv};

  v8f acc[3][4];                       // 3 projections x 4 N-tiles of 16
#pragma unroll
  for (int w = 0; w < 3; ++w)
#pragma unroll
    for (int nt = 0; nt < 4; ++nt)
      acc[w][nt] = (v8f){0.f,0.f,0.f,0.f,0.f,0.f,0.f,0.f};

  for (int d0 = 0; d0 < DMODEL; d0 += 32) {
    __syncthreads();
    // stage W chunk [32,64] f32 -> transposed f16 [64,32] in LDS
#pragma unroll
    for (int w = 0; w < 3; ++w)
      for (int idx = tid; idx < 32 * 64; idx += 256) {
        int d = idx >> 6, n = idx & 63;           // coalesced over n
        whT[w][n][d] = (_Float16)W[w][(size_t)(d0 + d) * 64 + n];
      }
    __syncthreads();

    // A operand: x rows, contiguous run {16*hlf .. +15}
    const float* xr = x + (size_t)(rowBase + ln) * DMODEL + d0;
    const int r0 = hlf * 16, r1 = hlf * 16 + 8;
    v4f p0 = *(const v4f*)(xr + r0);
    v4f p1 = *(const v4f*)(xr + r0 + 4);
    v4f p2 = *(const v4f*)(xr + r1);
    v4f p3 = *(const v4f*)(xr + r1 + 4);
    v16h a;
#pragma unroll
    for (int i = 0; i < 4; ++i) {
      a[i]      = (_Float16)p0[i];
      a[4 + i]  = (_Float16)p1[i];
      a[8 + i]  = (_Float16)p2[i];
      a[12 + i] = (_Float16)p3[i];
    }

#pragma unroll
    for (int w = 0; w < 3; ++w) {
      // load all 4 B operands first (distinct regs), then 4 WMMAs
      v16h bm[4];
#pragma unroll
      for (int nt = 0; nt < 4; ++nt) {
        const _Float16* bp = &whT[w][nt * 16 + ln][16 * hlf];
        bm[nt] = cat16(*(const v8h*)(bp), *(const v8h*)(bp + 8));
      }
#pragma unroll
      for (int nt = 0; nt < 4; ++nt)
        acc[w][nt] = wmma_f16(a, bm[nt], acc[w][nt]);
    }
  }

  // epilogue: C layout -> lane l, vgpr r is element (M = r+8*hlf, N = ln)
#pragma unroll
  for (int nt = 0; nt < 4; ++nt)
#pragma unroll
    for (int r = 0; r < 8; ++r) {
      int row = rowBase + r + 8 * hlf;
      int col = nt * 16 + ln;
      qh[(size_t)row * 64 + col] = (_Float16)acc[0][nt][r];
      kh[(size_t)row * 64 + col] = (_Float16)acc[1][nt][r];
      int b = row >> 12, s = row & 4095;
      vt[((size_t)b * 64 + col) * S_LEN + s] = (_Float16)acc[2][nt][r];
    }
}

// ---------------------------------------------------------------- kernel 2
// Flash attention: one wave per 16-row query tile, streaming key chunks of
// 32 with online softmax; P re-laid out C->A via per-wave LDS staging.
__global__ __launch_bounds__(256)
void flash_attn_kernel(const _Float16* __restrict__ qh,
                       const _Float16* __restrict__ kh,
                       const _Float16* __restrict__ vt,
                       float* __restrict__ out)
{
  __shared__ _Float16 pstage[8][16][32];   // 1KB per wave

  const int tid  = threadIdx.x;
  const int wave = tid >> 5;
  const int lane = tid & 31;
  const int hlf  = lane >> 4;
  const int ln   = lane & 15;
  const int rowBase = blockIdx.x * 128 + wave * 16;
  const int b  = rowBase >> 12;        // batch (tiles never straddle batches)
  const int q0 = rowBase & 4095;       // query position within sequence
  const float scale = 0.125f;          // 1/sqrt(64)
  const float NEG = -1e30f;

  // preload Q as two A operands (h = 0..31 and 32..63)
  const _Float16* qr = qh + (size_t)(rowBase + ln) * 64;
  v16h aq0 = cat16(*(const v8h*)(qr + 16 * hlf),
                   *(const v8h*)(qr + 16 * hlf + 8));
  v16h aq1 = cat16(*(const v8h*)(qr + 32 + 16 * hlf),
                   *(const v8h*)(qr + 40 + 16 * hlf));

  float m[8], l[8];
  v8f o[4];
#pragma unroll
  for (int r = 0; r < 8; ++r) { m[r] = NEG; l[r] = 0.f; }
#pragma unroll
  for (int n = 0; n < 4; ++n)
    o[n] = (v8f){0.f,0.f,0.f,0.f,0.f,0.f,0.f,0.f};

  const _Float16* vtb = vt + (size_t)b * 64 * S_LEN;
  const _Float16* khb = kh + (size_t)b * 4096 * 64;

  for (int kc = 0; kc < q0 + 16; kc += 32) {     // wave-uniform bound
    if (kc + 32 < q0 + 16)
      __builtin_prefetch(khb + (size_t)(kc + 32 + ln) * 64, 0, 1);

    // ---- scores for two 16-key tiles: load all K operands, then 4 WMMAs
    const _Float16* kr0 = khb + (size_t)(kc + ln) * 64;
    const _Float16* kr1 = khb + (size_t)(kc + 16 + ln) * 64;
    v16h bk00 = cat16(*(const v8h*)(kr0 + 16 * hlf),
                      *(const v8h*)(kr0 + 16 * hlf + 8));
    v16h bk01 = cat16(*(const v8h*)(kr0 + 32 + 16 * hlf),
                      *(const v8h*)(kr0 + 40 + 16 * hlf));
    v16h bk10 = cat16(*(const v8h*)(kr1 + 16 * hlf),
                      *(const v8h*)(kr1 + 16 * hlf + 8));
    v16h bk11 = cat16(*(const v8h*)(kr1 + 32 + 16 * hlf),
                      *(const v8h*)(kr1 + 40 + 16 * hlf));
    v8f cz = (v8f){0.f,0.f,0.f,0.f,0.f,0.f,0.f,0.f};
    v8f c0 = wmma_f16(aq0, bk00, cz);
    v8f c1 = wmma_f16(aq0, bk10, cz);
    c0     = wmma_f16(aq1, bk01, c0);
    c1     = wmma_f16(aq1, bk11, c1);

    v8f sc[2];
#pragma unroll
    for (int r = 0; r < 8; ++r) {
      int row = q0 + r + 8 * hlf;
      sc[0][r] = ((kc + ln)      > row) ? NEG : c0[r] * scale;  // causal mask
      sc[1][r] = ((kc + 16 + ln) > row) ? NEG : c1[r] * scale;
    }

    // ---- online softmax update + stage P into LDS (C->A re-layout)
#pragma unroll
    for (int r = 0; r < 8; ++r) {
      float cm = row_max16(fmaxf(sc[0][r], sc[1][r]));
      float mn = fmaxf(m[r], cm);
      float al = __expf(m[r] - mn);
      float e0 = __expf(sc[0][r] - mn);
      float e1 = __expf(sc[1][r] - mn);
      l[r] = l[r] * al + row_sum16(e0 + e1);
      m[r] = mn;
      o[0][r] *= al; o[1][r] *= al; o[2][r] *= al; o[3][r] *= al;
      pstage[wave][r + 8 * hlf][ln]      = (_Float16)e0;
      pstage[wave][r + 8 * hlf][16 + ln] = (_Float16)e1;
    }

    // per-wave lockstep: DS stores above precede DS loads below in order
    const _Float16* pr = &pstage[wave][ln][0];
    v16h ap = cat16(*(const v8h*)(pr + 16 * hlf),
                    *(const v8h*)(pr + 16 * hlf + 8));

    // ---- O += P * V: load all 4 V operands first, then 4 WMMAs
    v16h bv[4];
#pragma unroll
    for (int n = 0; n < 4; ++n) {
      const _Float16* vr = vtb + (size_t)(16 * n + ln) * S_LEN + kc + 16 * hlf;
      bv[n] = cat16(*(const v8h*)(vr), *(const v8h*)(vr + 8));
    }
#pragma unroll
    for (int n = 0; n < 4; ++n)
      o[n] = wmma_f16(ap, bv[n], o[n]);
  }

  // ---- normalize and store fp32 output [B,S,64]
#pragma unroll
  for (int r = 0; r < 8; ++r) {
    float inv = 1.0f / l[r];
    int row = rowBase + r + 8 * hlf;
#pragma unroll
    for (int n = 0; n < 4; ++n)
      out[(size_t)row * 64 + 16 * n + ln] = o[n][r] * inv;
  }
}

// ---------------------------------------------------------------- launcher
extern "C" void kernel_launch(void* const* d_in, const int* in_sizes, int n_in,
                              void* d_out, int out_size, void* d_ws, size_t ws_size,
                              hipStream_t stream) {
  (void)in_sizes; (void)n_in; (void)out_size; (void)ws_size;
  const float* x  = (const float*)d_in[0];
  const float* Wq = (const float*)d_in[1];
  const float* Wk = (const float*)d_in[2];
  const float* Wv = (const float*)d_in[3];
  float* out = (float*)d_out;

  _Float16* qh = (_Float16*)d_ws;                     // [NROWS,64]
  _Float16* kh = qh + (size_t)NROWS * DHEAD;          // [NROWS,64]
  _Float16* vt = kh + (size_t)NROWS * DHEAD;          // [B,64,S]

  proj_qkv_kernel<<<NROWS / 128, 256, 0, stream>>>(x, Wq, Wk, Wv, qh, kh, vt);
  flash_attn_kernel<<<NROWS / 128, 256, 0, stream>>>(qh, kh, vt, out);
}
